// XFedFormer_19447611916810
// MI455X (gfx1250) — matrix operand order repaired and checked by reading, not compiled, not run on hardware
//
#include <hip/hip_runtime.h>
#include <math.h>

// ---------------------------------------------------------------------------
// CDNA5 WMMA types / helpers
// ---------------------------------------------------------------------------
typedef __attribute__((ext_vector_type(16))) _Float16 v16h;
typedef __attribute__((ext_vector_type(8)))  float    v8f;

union AFrag { v16h v; _Float16 e[16]; };
union CFrag { v8f  v; float    e[8];  };
union PackH2 { _Float16 h[2]; unsigned int u; };

__device__ inline v8f wmma32(const AFrag& a, const AFrag& b, const CFrag& c) {
  return __builtin_amdgcn_wmma_f32_16x16x32_f16(false, a.v, false, b.v,
                                                (short)0, c.v, false, false);
}

// Fragment-major LDS tiles: each 16x32(A) / 32x16(B) sub-tile stored as
// [lane 0..31][j 0..15] halfs -> one contiguous 32B v16h load (2x ds_load_b128).
// A 16x32 (MxK) inverse: lane = (m&15) + ((k&8)<<1) ; j = (k&7) + ((k&16)>>1)
// B 32x16 (KxN) inverse: lane = (n&15) + ((k&8)<<1) ; j = (k&7) + ((k&16)>>1)
__device__ inline void frag_load(AFrag& f, const _Float16* tileBase, int lane) {
  f.v = *(const v16h*)(tileBase + (lane << 4));
}

// ---------------------------------------------------------------------------
// WMMA GEMM: C[M,N] (+)= epilogue(A[M,K] @ W[K,N] + bias)
// 64x128 block tile, 8 waves (each: 16x64 out = 4 wmma / k-step), BK=32.
// Double-buffered LDS, register-prefetch software pipeline.
// act: 0=none 1=gelu(exact) 2=relu. rowscale optional per-row multiplier.
// Requires M%64==0, N%128==0, K%32==0 (true for every call here).
// ---------------------------------------------------------------------------
__global__ __launch_bounds__(256) void gemm_kernel(
    const float* __restrict__ A, const float* __restrict__ W,
    const float* __restrict__ bias, const float* __restrict__ rowscale,
    int rs_stride, float* __restrict__ C,
    int M, int N, int K, int act, int accum)
{
  __shared__ alignas(32) _Float16 sA[2][4 * 32 * 16];  // [buf][mt][lane][j]
  __shared__ alignas(32) _Float16 sB[2][8 * 32 * 16];  // [buf][nt][lane][j]

  int tid  = threadIdx.x;
  int wave = tid >> 5;
  int lane = tid & 31;
  int m0 = blockIdx.y * 64;
  int n0 = blockIdx.x * 128;
  int mt  = wave & 3;            // wave's 16-row tile within 64
  int nt0 = (wave >> 2) * 4;     // wave's four 16-col tiles within 128

  // ---- precomputed staging addressing (loop-invariant) ----
  // A: 64x32 tile = 1024 pairs, 4 per thread (rows ar+16*it, col pair ac)
  int ar = tid >> 4;
  int ac = (tid & 15) << 1;
  int aoff = ((ar + ((ac & 8) << 1)) << 4) + (ac & 7) + ((ac & 16) >> 1);
  const float* pA = A + (size_t)(m0 + ar) * K + ac;

  // B: 32x128 tile = 2048 pairs, 8 per thread (col bn, k pair bk0+4*it)
  int bn  = tid & 127;
  int bk0 = (tid >> 7) << 1;
  int boff[8];
#pragma unroll
  for (int it = 0; it < 8; ++it) {
    int kk = bk0 + it * 4;
    boff[it] = (((bn >> 4) * 32 + (bn & 15) + ((kk & 8) << 1)) << 4) + (kk & 7) + ((kk & 16) >> 1);
  }
  const float* pB = W + (size_t)bk0 * N + n0 + bn;

  float2 ra[4];
  float  rb[8][2];

  auto gload = [&](int k0) {
    const float* a = pA + k0;
#pragma unroll
    for (int it = 0; it < 4; ++it)
      ra[it] = *(const float2*)(a + (size_t)(it * 16) * K);
    const float* b = pB + (size_t)k0 * N;
#pragma unroll
    for (int it = 0; it < 8; ++it) {
      rb[it][0] = b[(size_t)(it * 4) * N];
      rb[it][1] = b[(size_t)(it * 4) * N + N];
    }
  };
  auto sstore = [&](int buf) {
    _Float16* dA = sA[buf];
    _Float16* dB = sB[buf];
#pragma unroll
    for (int it = 0; it < 4; ++it) {
      PackH2 pk; pk.h[0] = (_Float16)ra[it].x; pk.h[1] = (_Float16)ra[it].y;
      *(unsigned int*)(dA + (it << 9) + aoff) = pk.u;
    }
#pragma unroll
    for (int it = 0; it < 8; ++it) {
      PackH2 pk; pk.h[0] = (_Float16)rb[it][0]; pk.h[1] = (_Float16)rb[it][1];
      *(unsigned int*)(dB + boff[it]) = pk.u;
    }
  };

  CFrag acc[4];
#pragma unroll
  for (int t = 0; t < 4; ++t)
#pragma unroll
    for (int i = 0; i < 8; ++i) acc[t].e[i] = 0.f;

  gload(0);
  int buf = 0;
  for (int k0 = 0; k0 < K; k0 += 32) {
    sstore(buf);
    __syncthreads();
    if (k0 + 32 < K) gload(k0 + 32);   // prefetch overlaps the WMMAs below
    AFrag af;
    frag_load(af, sA[buf] + (mt << 9), lane);
#pragma unroll
    for (int t = 0; t < 4; ++t) {
      AFrag bf;
      frag_load(bf, sB[buf] + ((nt0 + t) << 9), lane);
      acc[t].v = wmma32(af, bf, acc[t]);
    }
    __syncthreads();
    buf ^= 1;
  }

  // epilogue: lane holds column n=lane%16, VGPR r holds row r(+8)
  int n  = lane & 15;
  int mo = (lane < 16) ? 0 : 8;
#pragma unroll
  for (int r = 0; r < 8; ++r) {
    int gm = m0 + mt * 16 + mo + r;
    float rsv = rowscale ? rowscale[(size_t)gm * rs_stride] : 1.f;
#pragma unroll
    for (int t = 0; t < 4; ++t) {
      int gn = n0 + (nt0 + t) * 16 + n;
      float v = acc[t].e[r];
      if (bias) v += bias[gn];
      if (act == 1)      v = 0.5f * v * (1.0f + erff(v * 0.70710678118654752f));
      else if (act == 2) v = fmaxf(v, 0.f);
      v *= rsv;
      size_t off = (size_t)gm * N + gn;
      if (accum) C[off] += v; else C[off] = v;
    }
  }
}

// ---------------------------------------------------------------------------
// Flash attention: one workgroup per (64 q-rows, head, batch).
// qkv: [B,T,3*D] fp32 (q|k|v per token). out: [B,T,D] fp32.
// Fragment-major LDS; K/V staging double-buffered with register prefetch.
// ---------------------------------------------------------------------------
__global__ __launch_bounds__(256) void attn_kernel(
    const float* __restrict__ qkv, float* __restrict__ out,
    int T, int Dm, int dh, float scale)
{
  __shared__ alignas(32) _Float16 sQF[4 * 2 * 32 * 16];     // [mt][kchunk][lane][j]
  __shared__ alignas(32) _Float16 sKF[2][2 * 2 * 32 * 16];  // [buf][kchunk][nt][lane][j]
  __shared__ alignas(32) _Float16 sVF[2][4 * 32 * 16];      // [buf][nt][lane][j]
  __shared__ alignas(32) _Float16 sPF[4 * 32 * 16];         // [mt][lane][j]
  __shared__ float sS[64][33];
  __shared__ float sMx[64], sL[64], sCf[64];

  int b = blockIdx.z, h = blockIdx.y;
  int q0 = blockIdx.x * 64;
  int tid = threadIdx.x, wave = tid >> 5, lane = tid & 31;
  int mt   = wave & 3;          // 16-row tile
  int snt  = wave >> 2;         // S col tile (0/1 of 32 keys)
  int ont0 = (wave >> 2) * 2;   // O col tiles (of 64 = dh)

  size_t row3 = 3 * (size_t)Dm;
  const float* base = qkv + (size_t)b * T * row3;

  // ---- stage Q once (two-phase: 8 loads then 8 stores) ----
  {
    int c0 = (tid & 31) << 1;
    float2 rq[8];
#pragma unroll
    for (int it = 0; it < 8; ++it) {
      int r = (tid >> 5) + it * 8;
      rq[it] = *(const float2*)(base + (size_t)(q0 + r) * row3 + h * dh + c0);
    }
    int chunk = c0 >> 5;
    int cc = c0 & 31;
    int lj = (((cc & 8) << 1) << 4) + (cc & 7) + ((cc & 16) >> 1);
#pragma unroll
    for (int it = 0; it < 8; ++it) {
      int r = (tid >> 5) + it * 8;
      PackH2 pk; pk.h[0] = (_Float16)rq[it].x; pk.h[1] = (_Float16)rq[it].y;
      *(unsigned int*)(sQF + ((((r >> 4) * 2 + chunk) * 32 + (r & 15)) << 4) + lj) = pk.u;
    }
  }
  if (tid < 64) { sMx[tid] = -3.0e38f; sL[tid] = 0.f; }

  // ---- precomputed K/V staging addressing ----
  int kkd = (tid & 31) << 1;                  // dh pair (const)
  int kn0 = tid >> 5;                         // key base, +8*it
  int kchunk = kkd >> 5, kkk = kkd & 31;
  int koff[4];
#pragma unroll
  for (int it = 0; it < 4; ++it) {
    int nq = kn0 + it * 8;
    koff[it] = (((kchunk * 2 + (nq >> 4)) * 32 + (nq & 15) + ((kkk & 8) << 1)) << 4) + (kkk & 7) + ((kkk & 16) >> 1);
  }
  int vkd = tid & 63;                         // dh col (const)
  int vk0 = (tid >> 6) << 1;                  // key pair base, +8*it
  int voff[4];
#pragma unroll
  for (int it = 0; it < 4; ++it) {
    int kp = vk0 + it * 8;
    voff[it] = (((vkd >> 4) * 32 + (vkd & 15) + ((kp & 8) << 1)) << 4) + (kp & 7) + ((kp & 16) >> 1);
  }

  float2 rk[4];
  float  rv[4][2];
  auto kvload = [&](int j0) {
#pragma unroll
    for (int it = 0; it < 4; ++it)
      rk[it] = *(const float2*)(base + (size_t)(j0 + kn0 + it * 8) * row3 + Dm + h * dh + kkd);
#pragma unroll
    for (int it = 0; it < 4; ++it) {
      const float* pv = base + (size_t)(j0 + vk0 + it * 8) * row3 + 2 * Dm + h * dh + vkd;
      rv[it][0] = pv[0];
      rv[it][1] = pv[row3];
    }
  };
  auto kvstore = [&](int buf) {
#pragma unroll
    for (int it = 0; it < 4; ++it) {
      PackH2 pk; pk.h[0] = (_Float16)rk[it].x; pk.h[1] = (_Float16)rk[it].y;
      *(unsigned int*)(sKF[buf] + koff[it]) = pk.u;
    }
#pragma unroll
    for (int it = 0; it < 4; ++it) {
      PackH2 pk; pk.h[0] = (_Float16)rv[it][0]; pk.h[1] = (_Float16)rv[it][1];
      *(unsigned int*)(sVF[buf] + voff[it]) = pk.u;
    }
  };

  CFrag o0, o1;
#pragma unroll
  for (int i = 0; i < 8; ++i) { o0.e[i] = 0.f; o1.e[i] = 0.f; }

  kvload(0);
  int buf = 0;
  __syncthreads();

  for (int j0 = 0; j0 < T; j0 += 32) {
    kvstore(buf);
    __syncthreads();
    if (j0 + 32 < T) kvload(j0 + 32);   // prefetch overlaps all compute below

    // S tile = Q(16x64) @ K^T(64x16) per wave, two 32-K chunks
    CFrag sacc;
#pragma unroll
    for (int i = 0; i < 8; ++i) sacc.e[i] = 0.f;
#pragma unroll
    for (int chunk = 0; chunk < 2; ++chunk) {
      AFrag af, bf;
      frag_load(af, sQF + (((mt * 2 + chunk) << 5) << 4), lane);
      frag_load(bf, sKF[buf] + (((chunk * 2 + snt) << 5) << 4), lane);
      sacc.v = wmma32(af, bf, sacc);
    }
    {
      int n  = lane & 15;
      int mo = (lane < 16) ? 0 : 8;
#pragma unroll
      for (int r = 0; r < 8; ++r)
        sS[mt * 16 + mo + r][snt * 16 + n] = sacc.e[r] * scale;
    }
    __syncthreads();

    // online softmax over this 32-key block (one thread per row);
    // write P directly in A-fragment-major order (packed pairs)
    if (tid < 64) {
      float mold = sMx[tid];
      float mx = mold;
#pragma unroll
      for (int c = 0; c < 32; ++c) mx = fmaxf(mx, sS[tid][c]);
      float cf = __expf(mold - mx);
      float l = sL[tid] * cf;
      _Float16* prow = sPF + (((tid >> 4) * 32) << 4);
      int m15 = tid & 15;
#pragma unroll
      for (int c0 = 0; c0 < 32; c0 += 2) {
        float p0 = __expf(sS[tid][c0]     - mx);
        float p1 = __expf(sS[tid][c0 + 1] - mx);
        l += p0 + p1;
        int ll = m15 + ((c0 & 8) << 1);
        int jj = (c0 & 7) + ((c0 & 16) >> 1);
        PackH2 pk; pk.h[0] = (_Float16)p0; pk.h[1] = (_Float16)p1;
        *(unsigned int*)(prow + (ll << 4) + jj) = pk.u;
      }
      sMx[tid] = mx; sL[tid] = l; sCf[tid] = cf;
    }
    __syncthreads();

    // rescale O accumulators by per-row correction factor
    {
      int mo = (lane < 16) ? 0 : 8;
#pragma unroll
      for (int r = 0; r < 8; ++r) {
        float cf = sCf[mt * 16 + mo + r];
        o0.e[r] *= cf; o1.e[r] *= cf;
      }
    }
    // O += P(16x32) @ V(32x16,16)
    {
      AFrag pf, vf0, vf1;
      frag_load(pf,  sPF + ((mt << 5) << 4),              lane);
      frag_load(vf0, sVF[buf] + ((ont0 << 5) << 4),       lane);
      frag_load(vf1, sVF[buf] + (((ont0 + 1) << 5) << 4), lane);
      o0.v = wmma32(pf, vf0, o0);
      o1.v = wmma32(pf, vf1, o1);
    }
    __syncthreads();
    buf ^= 1;
  }

  // finalize: divide by softmax denominator, store O
  int n  = lane & 15;
  int mo = (lane < 16) ? 0 : 8;
#pragma unroll
  for (int r = 0; r < 8; ++r) {
    int row = mt * 16 + mo + r;
    float inv = 1.f / sL[row];
    size_t o = ((size_t)b * T + q0 + row) * Dm + h * dh;
    out[o + ont0 * 16 + n]      = o0.e[r] * inv;
    out[o + ont0 * 16 + 16 + n] = o1.e[r] * inv;
  }
}

// ---------------------------------------------------------------------------
// Elementwise / small kernels
// ---------------------------------------------------------------------------
__global__ void decomp_kernel(const float* __restrict__ x, float* __restrict__ resid,
                              float* __restrict__ trendl, int B, int T, int C) {
  int idx = blockIdx.x * blockDim.x + threadIdx.x;
  if (idx >= B * T * C) return;
  int c = idx % C;
  int t = (idx / C) % T;
  int b = idx / (C * T);
  float s = 0.f;
#pragma unroll
  for (int dt = -3; dt <= 3; ++dt) {
    int tt = t + dt;
    if (tt >= 0 && tt < T) s += x[((size_t)b * T + tt) * C + c];
  }
  float tr = s * (1.f / 7.f);
  resid[idx] = x[idx] - tr;
  if (t == T - 1) trendl[b * C + c] = tr;
}

__global__ void addpos_kernel(float* __restrict__ z, const float* __restrict__ pos,
                              int Ntok, int Dm, int T) {
  int idx = blockIdx.x * blockDim.x + threadIdx.x;
  if (idx >= Ntok * Dm) return;
  int d = idx % Dm;
  int t = (idx / Dm) % T;
  z[idx] += pos[(size_t)t * Dm + d];
}

// LayerNorm over D=512 (2 elems/thread, 256 threads/row); optional residual add.
__global__ __launch_bounds__(256) void ln_kernel(
    const float* __restrict__ in, const float* __restrict__ addin,
    const float* __restrict__ g, const float* __restrict__ bt,
    float* __restrict__ out, int Dm)
{
  __shared__ float red[256];
  int row = blockIdx.x, tid = threadIdx.x;
  const float* xr = in + (size_t)row * Dm;
  float v0 = xr[tid], v1 = xr[tid + 256];
  if (addin) {
    const float* ar = addin + (size_t)row * Dm;
    v0 += ar[tid]; v1 += ar[tid + 256];
  }
  red[tid] = v0 + v1;
  __syncthreads();
  for (int s = 128; s > 0; s >>= 1) { if (tid < s) red[tid] += red[tid + s]; __syncthreads(); }
  float mean = red[0] / (float)Dm;
  __syncthreads();
  float d0 = v0 - mean, d1 = v1 - mean;
  red[tid] = d0 * d0 + d1 * d1;
  __syncthreads();
  for (int s = 128; s > 0; s >>= 1) { if (tid < s) red[tid] += red[tid + s]; __syncthreads(); }
  float rs = rsqrtf(red[0] / (float)Dm + 1e-5f);
  out[(size_t)row * Dm + tid]       = d0 * rs * g[tid] + bt[tid];
  out[(size_t)row * Dm + tid + 256] = d1 * rs * g[tid + 256] + bt[tid + 256];
}

// tiny GEMM for router logits (N=4)
__global__ void logits_kernel(const float* __restrict__ A, const float* __restrict__ W,
                              const float* __restrict__ bias, float* __restrict__ C,
                              int M, int N, int K) {
  int idx = blockIdx.x * blockDim.x + threadIdx.x;
  if (idx >= M * N) return;
  int m = idx / N, n = idx % N;
  float acc = bias[n];
  for (int k = 0; k < K; ++k) acc += A[(size_t)m * K + k] * W[(size_t)k * N + n];
  C[idx] = acc;
}

// top-2 softmax gating -> dense [N,4] weights
__global__ void gate_kernel(const float* __restrict__ logits, float* __restrict__ gw, int Ntok) {
  int nn = blockIdx.x * blockDim.x + threadIdx.x;
  if (nn >= Ntok) return;
  const float* lg = logits + (size_t)nn * 4;
  float best = lg[0]; int bi = 0;
#pragma unroll
  for (int e = 1; e < 4; ++e) if (lg[e] > best) { best = lg[e]; bi = e; }
  float best2 = -3.0e38f; int bi2 = 0;
#pragma unroll
  for (int e = 0; e < 4; ++e) if (e != bi && lg[e] > best2) { best2 = lg[e]; bi2 = e; }
  float e2 = __expf(best2 - best);
  float inv = 1.f / (1.f + e2);
  float* o = gw + (size_t)nn * 4;
  o[0] = 0.f; o[1] = 0.f; o[2] = 0.f; o[3] = 0.f;
  o[bi]  = inv;
  o[bi2] = e2 * inv;
}

// forecast = z[:, T-1, :] @ Wdec + bdec + trend_last
__global__ void decode_kernel(const float* __restrict__ Z, const float* __restrict__ Wdec,
                              const float* __restrict__ bdec, const float* __restrict__ trendl,
                              float* __restrict__ out, int B, int T, int Dm, int NR) {
  int idx = blockIdx.x * blockDim.x + threadIdx.x;
  if (idx >= B * NR) return;
  int b = idx / NR, r = idx % NR;
  const float* zr = Z + ((size_t)b * T + (T - 1)) * Dm;
  float acc = bdec[r];
  for (int d = 0; d < Dm; ++d) acc += zr[d] * Wdec[(size_t)d * NR + r];
  out[idx] = acc + trendl[idx];
}

// ---------------------------------------------------------------------------
// Orchestration
// ---------------------------------------------------------------------------
extern "C" void kernel_launch(void* const* d_in, const int* in_sizes, int n_in,
                              void* d_out, int out_size, void* d_ws, size_t ws_size,
                              hipStream_t stream) {
  (void)in_sizes; (void)n_in; (void)out_size; (void)ws_size;

  const int B = 8, T = 1024, IN = 256, Dm = 512, Hh = 8, dh = 64;
  const int Ntok = B * T;               // 8192
  const int DFF = 2048, NE = 4, NR = 256;

  const float* x       = (const float*)d_in[0];
  const float* Wproj   = (const float*)d_in[1];
  const float* bproj   = (const float*)d_in[2];
  const float* pos_enc = (const float*)d_in[3];
  const float* Wqkv    = (const float*)d_in[4];
  const float* bqkv    = (const float*)d_in[5];
  const float* Wo      = (const float*)d_in[6];
  const float* bo      = (const float*)d_in[7];
  const float* W1      = (const float*)d_in[8];
  const float* b1      = (const float*)d_in[9];
  const float* W2      = (const float*)d_in[10];
  const float* b2      = (const float*)d_in[11];
  const float* ln1_g   = (const float*)d_in[12];
  const float* ln1_b   = (const float*)d_in[13];
  const float* ln2_g   = (const float*)d_in[14];
  const float* ln2_b   = (const float*)d_in[15];
  const float* lnf_g   = (const float*)d_in[16];
  const float* lnf_b   = (const float*)d_in[17];
  const float* Wg      = (const float*)d_in[18];
  const float* bg      = (const float*)d_in[19];
  const float* We1     = (const float*)d_in[20];
  const float* be1     = (const float*)d_in[21];
  const float* We2     = (const float*)d_in[22];
  const float* be2     = (const float*)d_in[23];
  const float* moe_g   = (const float*)d_in[24];
  const float* moe_b   = (const float*)d_in[25];
  const float* Wdec    = (const float*)d_in[26];
  const float* bdec    = (const float*)d_in[27];

  // workspace layout (fp32)
  float* ws     = (float*)d_ws;
  float* RESID  = ws;                                  // 8192*256
  float* TRENDL = RESID + (size_t)Ntok * IN;           // 8*256
  float* Z      = TRENDL + (size_t)B * NR;             // 8192*512
  float* T1     = Z + (size_t)Ntok * Dm;               // 8192*512
  float* BIG    = T1 + (size_t)Ntok * Dm;              // 8192*2048
  float* GATE   = BIG + (size_t)Ntok * DFF;            // 8192*4
  float* GW     = GATE + (size_t)Ntok * NE;            // 8192*4

  auto gemm = [&](const float* A, const float* W, const float* bias,
                  const float* rowscale, int rss, float* C,
                  int M, int N, int K, int act, int accum) {
    dim3 grid(N / 128, M / 64);
    gemm_kernel<<<grid, 256, 0, stream>>>(A, W, bias, rowscale, rss, C, M, N, K, act, accum);
  };

  // 1) seasonal-trend decomposition
  decomp_kernel<<<(B * T * IN + 255) / 256, 256, 0, stream>>>(x, RESID, TRENDL, B, T, IN);

  // 2) input projection + positional encoding
  gemm(RESID, Wproj, bproj, nullptr, 0, Z, Ntok, Dm, IN, 0, 0);
  addpos_kernel<<<(Ntok * Dm + 255) / 256, 256, 0, stream>>>(Z, pos_enc, Ntok, Dm, T);

  // 3) transformer encoder layers (post-LN, gelu)
  for (int l = 0; l < 2; ++l) {
    const float* Wqkv_l = Wqkv + (size_t)l * Dm * 3 * Dm;
    const float* bqkv_l = bqkv + (size_t)l * 3 * Dm;
    const float* Wo_l   = Wo + (size_t)l * Dm * Dm;
    const float* bo_l   = bo + (size_t)l * Dm;
    const float* W1_l   = W1 + (size_t)l * Dm * DFF;
    const float* b1_l   = b1 + (size_t)l * DFF;
    const float* W2_l   = W2 + (size_t)l * DFF * Dm;
    const float* b2_l   = b2 + (size_t)l * Dm;

    gemm(Z, Wqkv_l, bqkv_l, nullptr, 0, BIG, Ntok, 3 * Dm, Dm, 0, 0);   // QKV
    attn_kernel<<<dim3(T / 64, Hh, B), 256, 0, stream>>>(BIG, T1, T, Dm, dh, 0.125f);
    gemm(T1, Wo_l, bo_l, nullptr, 0, BIG, Ntok, Dm, Dm, 0, 0);          // attn out proj
    ln_kernel<<<Ntok, 256, 0, stream>>>(BIG, Z, ln1_g + l * Dm, ln1_b + l * Dm, Z, Dm);
    gemm(Z, W1_l, b1_l, nullptr, 0, BIG, Ntok, DFF, Dm, 1, 0);          // FF1 + gelu
    gemm(BIG, W2_l, b2_l, nullptr, 0, T1, Ntok, Dm, DFF, 0, 0);         // FF2
    ln_kernel<<<Ntok, 256, 0, stream>>>(T1, Z, ln2_g + l * Dm, ln2_b + l * Dm, Z, Dm);
  }

  // 4) final encoder LN (in-place safe: row fully read before write)
  ln_kernel<<<Ntok, 256, 0, stream>>>(Z, nullptr, lnf_g, lnf_b, Z, Dm);

  // 5) MoE: router + top-2 gating + weighted expert mix
  logits_kernel<<<(Ntok * NE + 255) / 256, 256, 0, stream>>>(Z, Wg, bg, GATE, Ntok, NE, Dm);
  gate_kernel<<<(Ntok + 255) / 256, 256, 0, stream>>>(GATE, GW, Ntok);
  for (int e = 0; e < NE; ++e) {
    const float* We1_e = We1 + (size_t)e * Dm * 2 * Dm;
    const float* be1_e = be1 + (size_t)e * 2 * Dm;
    const float* We2_e = We2 + (size_t)e * 2 * Dm * Dm;
    const float* be2_e = be2 + (size_t)e * Dm;
    gemm(Z, We1_e, be1_e, nullptr, 0, BIG, Ntok, 2 * Dm, Dm, 2, 0);            // relu
    gemm(BIG, We2_e, be2_e, GW + e, NE, T1, Ntok, Dm, 2 * Dm, 0, e > 0);       // += gate*expert
  }
  ln_kernel<<<Ntok, 256, 0, stream>>>(T1, Z, moe_g, moe_b, Z, Dm);

  // 6) decode head + trend add
  decode_kernel<<<(B * NR + 255) / 256, 256, 0, stream>>>(Z, Wdec, bdec, TRENDL,
                                                          (float*)d_out, B, T, Dm, NR);
}
